// PCI5Attention_59760174956722
// MI455X (gfx1250) — compile-verified, matrix operand order checked by reading, not run on hardware
//
#include <hip/hip_runtime.h>
#include <hip/hip_bf16.h>

// ---------------------------------------------------------------------------
// PCI5Attention for MI455X (gfx1250): bf16 WMMA pipeline, double-buffered LDS
//   1) Q = rope(X@Wq^T + bq), K = rope(X@Wk^T + bk), V = X@Wv^T + bv  (bf16)
//   2) S = (Q K^T) / sqrt(128), causal masked                         (f32)
//   3) P = 2 * softmax(S)                                             (bf16)
//   4) O = P @ V                                                      (bf16)
//   5) out = O @ Wfc^T + bfc                                          (f32)
// bf16-source tiles use GLOBAL_LOAD_ASYNC_TO_LDS (ASYNCcnt) via inline asm;
// f32-source tiles are register-staged and converted to bf16 at commit.
// ---------------------------------------------------------------------------

typedef __bf16 v16bf __attribute__((ext_vector_type(16)));
typedef float  v8f   __attribute__((ext_vector_type(8)));

constexpr int BM = 128, BN = 128, BK = 32;
constexpr int LDSP = BK + 8;           // padded k-stride: 80 bytes/row (16B aligned)
constexpr int Tdim = 2048;
constexpr int Ddim = 2048;
constexpr int Bdim = 2;

struct alignas(16) U4 { unsigned int x, y, z, w; };
struct alignas(16) F4 { float x, y, z, w; };
struct alignas(8)  H4 { unsigned short x, y, z, w; };

union FragBF { v16bf v; U4 u[2]; };

// --- CDNA5 async global->LDS via inline asm (device pass only) -------------
#if defined(__gfx1250__)
#define K_ASYNC 1
#else
#define K_ASYNC 0
#endif

__device__ __forceinline__ void async_copy_b128(const void* g, void* l) {
#if K_ASYNC
  // VDST = wave-relative LDS byte address (low 32 bits of generic LDS pointer,
  // per ISA 10.2 aperture rules); VADDR = 64-bit global address (VGPR pair).
  unsigned int lds_off = (unsigned int)(unsigned long long)l;
  asm volatile("global_load_async_to_lds_b128 %0, %1, off"
               :: "v"(lds_off), "v"((unsigned long long)g)
               : "memory");
#else
  (void)g; (void)l;
#endif
}

__device__ __forceinline__ void async_wait_all() {
#if K_ASYNC
#if __has_builtin(__builtin_amdgcn_s_wait_asynccnt)
  __builtin_amdgcn_s_wait_asynccnt(0);
#else
  asm volatile("s_wait_asynccnt 0x0" ::: "memory");
#endif
#endif
}

__device__ __forceinline__ unsigned short f32_to_bf16(float f) {
  unsigned int u = __float_as_uint(f);
  u += 0x7fffu + ((u >> 16) & 1u);     // round-to-nearest-even
  return (unsigned short)(u >> 16);
}

// EPI: 0 = bias + f32 store (FC)
//      1 = bias + RoPE(first 128 cols) + bf16 store (Q/K proj)
//      2 = scale 1/sqrt(128) + causal mask + f32 store (scores)
//      3 = bias + bf16 store (V proj)
//      4 = plain bf16 store (A@V)
template<bool A_BF16, bool B_BF16, bool B_NN, int EPI>
__global__ __launch_bounds__(256) void gemm_kernel(
    const void* __restrict__ Ap, const void* __restrict__ Bp,
    const float* __restrict__ bias, void* __restrict__ Cp,
    int K, int lda, int ldb, int ldc,
    long sA, long sB, long sC)
{
  __shared__ __align__(16) unsigned short Alds[2][BM * LDSP];
  __shared__ __align__(16) unsigned short Blds[2][BN * LDSP];

  const int tid  = threadIdx.x;
  const int lane = tid & 31;
  const int wid  = tid >> 5;
  const int wm   = wid & 3;            // 4 waves along M  -> 32-row strip
  const int wn   = wid >> 2;           // 2 waves along N  -> 64-col strip
  const int bm0  = blockIdx.y * BM;
  const int bn0  = blockIdx.x * BN;

  const char* Ab = (const char*)Ap + (size_t)blockIdx.z * (size_t)sA * (A_BF16 ? 2 : 4);
  const char* Bb = (const char*)Bp + (size_t)blockIdx.z * (size_t)sB * (B_BF16 ? 2 : 4);
  const size_t cElem = (EPI == 0 || EPI == 2) ? 4u : 2u;
  char* Cb = (char*)Cp + (size_t)blockIdx.z * (size_t)sC * cElem;

  if (EPI == 2) {
    // whole tile above the causal diagonal: fill with -inf surrogate, skip GEMM
    if (bn0 > bm0 + BM - 1) {
      float* C = (float*)Cb;
      for (int i = tid; i < BM * BN; i += 256) {
        int r = i >> 7, c = i & 127;
        C[(size_t)(bm0 + r) * ldc + bn0 + c] = -1e30f;
      }
      return;
    }
  }

  const unsigned short* Ag16 = (const unsigned short*)Ab;
  const float*          Ag32 = (const float*)Ab;
  const unsigned short* Bg16 = (const unsigned short*)Bb;
  const float*          Bg32 = (const float*)Bb;

  // staged registers (unused variants are optimized away)
  U4 rA[2]; F4 fA[4];
  U4 rB[2]; F4 fB[4];

  auto issueA = [&](int kt, int buf) {
    if (A_BF16) {
      #pragma unroll
      for (int v = 0; v < 2; v++) {
        int idx = tid + v * 256; int r = idx >> 2, c8 = idx & 3;
        const unsigned short* src = Ag16 + (size_t)(bm0 + r) * lda + kt + c8 * 8;
        if (K_ASYNC) async_copy_b128(src, &Alds[buf][r * LDSP + c8 * 8]);
        else         rA[v] = *(const U4*)src;
      }
    } else {
      #pragma unroll
      for (int v = 0; v < 4; v++) {
        int idx = tid + v * 256; int r = idx >> 3, c4 = idx & 7;
        fA[v] = *(const F4*)(Ag32 + (size_t)(bm0 + r) * lda + kt + c4 * 4);
      }
    }
  };
  auto commitA = [&](int buf) {
    if (A_BF16) {
      if (!K_ASYNC) {
        #pragma unroll
        for (int v = 0; v < 2; v++) {
          int idx = tid + v * 256; int r = idx >> 2, c8 = idx & 3;
          *(U4*)&Alds[buf][r * LDSP + c8 * 8] = rA[v];
        }
      }
    } else {
      #pragma unroll
      for (int v = 0; v < 4; v++) {
        int idx = tid + v * 256; int r = idx >> 3, c4 = idx & 7;
        H4 h; h.x = f32_to_bf16(fA[v].x); h.y = f32_to_bf16(fA[v].y);
        h.z = f32_to_bf16(fA[v].z); h.w = f32_to_bf16(fA[v].w);
        *(H4*)&Alds[buf][r * LDSP + c4 * 4] = h;
      }
    }
  };
  auto issueB = [&](int kt, int buf) {
    if (!B_NN) {                        // B global is [N][K] (W^T style)
      if (B_BF16) {
        #pragma unroll
        for (int v = 0; v < 2; v++) {
          int idx = tid + v * 256; int r = idx >> 2, c8 = idx & 3;
          const unsigned short* src = Bg16 + (size_t)(bn0 + r) * ldb + kt + c8 * 8;
          if (K_ASYNC) async_copy_b128(src, &Blds[buf][r * LDSP + c8 * 8]);
          else         rB[v] = *(const U4*)src;
        }
      } else {
        #pragma unroll
        for (int v = 0; v < 4; v++) {
          int idx = tid + v * 256; int r = idx >> 3, c4 = idx & 7;
          fB[v] = *(const F4*)(Bg32 + (size_t)(bn0 + r) * ldb + kt + c4 * 4);
        }
      }
    } else {                            // B global is [K][N]: transpose at commit
      if (B_BF16) {
        #pragma unroll
        for (int v = 0; v < 2; v++) {
          int idx = tid + v * 256; int r = idx >> 4, c8 = idx & 15;
          rB[v] = *(const U4*)(Bg16 + (size_t)(kt + r) * ldb + bn0 + c8 * 8);
        }
      } else {
        #pragma unroll
        for (int v = 0; v < 4; v++) {
          int idx = tid + v * 256; int r = idx >> 5, c4 = idx & 31;
          fB[v] = *(const F4*)(Bg32 + (size_t)(kt + r) * ldb + bn0 + c4 * 4);
        }
      }
    }
  };
  auto commitB = [&](int buf) {
    if (!B_NN) {
      if (B_BF16) {
        if (!K_ASYNC) {
          #pragma unroll
          for (int v = 0; v < 2; v++) {
            int idx = tid + v * 256; int r = idx >> 2, c8 = idx & 3;
            *(U4*)&Blds[buf][r * LDSP + c8 * 8] = rB[v];
          }
        }
      } else {
        #pragma unroll
        for (int v = 0; v < 4; v++) {
          int idx = tid + v * 256; int r = idx >> 3, c4 = idx & 7;
          H4 h; h.x = f32_to_bf16(fB[v].x); h.y = f32_to_bf16(fB[v].y);
          h.z = f32_to_bf16(fB[v].z); h.w = f32_to_bf16(fB[v].w);
          *(H4*)&Blds[buf][r * LDSP + c4 * 4] = h;
        }
      }
    } else {
      if (B_BF16) {
        #pragma unroll
        for (int v = 0; v < 2; v++) {
          int idx = tid + v * 256; int r = idx >> 4, c8 = idx & 15;
          const unsigned short* e = (const unsigned short*)&rB[v];
          #pragma unroll
          for (int q = 0; q < 8; q++) Blds[buf][(c8 * 8 + q) * LDSP + r] = e[q];
        }
      } else {
        #pragma unroll
        for (int v = 0; v < 4; v++) {
          int idx = tid + v * 256; int r = idx >> 5, c4 = idx & 31;
          Blds[buf][(c4 * 4 + 0) * LDSP + r] = f32_to_bf16(fB[v].x);
          Blds[buf][(c4 * 4 + 1) * LDSP + r] = f32_to_bf16(fB[v].y);
          Blds[buf][(c4 * 4 + 2) * LDSP + r] = f32_to_bf16(fB[v].z);
          Blds[buf][(c4 * 4 + 3) * LDSP + r] = f32_to_bf16(fB[v].w);
        }
      }
    }
  };

  v8f acc[2][4];
  #pragma unroll
  for (int i = 0; i < 2; i++)
    #pragma unroll
    for (int j = 0; j < 4; j++)
      #pragma unroll
      for (int r = 0; r < 8; r++) acc[i][j][r] = 0.0f;

  const int nIter = K / BK;
  issueA(0, 0);
  issueB(0, 0);

  for (int it = 0; it < nIter; ++it) {
    const int cur = it & 1;
    commitA(cur);
    commitB(cur);
    if (K_ASYNC && (A_BF16 || (B_BF16 && !B_NN))) async_wait_all();
    __syncthreads();                       // buf[cur] ready; buf[cur^1] free
    if (it + 1 < nIter) {
      issueA((it + 1) * BK, cur ^ 1);      // overlaps with WMMA below
      issueB((it + 1) * BK, cur ^ 1);
      int ktp = (it + 2) * BK;             // L2 prefetch two tiles ahead
      if (ktp < K) {
        if (A_BF16) __builtin_prefetch(Ag16 + (size_t)(bm0 + (tid >> 2)) * lda + ktp, 0, 1);
        else        __builtin_prefetch(Ag32 + (size_t)(bm0 + (tid >> 3)) * lda + ktp, 0, 1);
      }
    }

    // ---- fragments: two 16B LDS reads per 16x32 fragment ----
    const int frow = lane & 15;
    const int k0   = (lane >> 4) * 8;
    FragBF afr[2], bfr[4];
    #pragma unroll
    for (int i = 0; i < 2; i++) {
      const unsigned short* base = &Alds[cur][(wm * 32 + i * 16 + frow) * LDSP + k0];
      afr[i].u[0] = *(const U4*)(base);
      afr[i].u[1] = *(const U4*)(base + 16);
    }
    #pragma unroll
    for (int j = 0; j < 4; j++) {
      const unsigned short* base = &Blds[cur][(wn * 64 + j * 16 + frow) * LDSP + k0];
      bfr[j].u[0] = *(const U4*)(base);
      bfr[j].u[1] = *(const U4*)(base + 16);
    }
    #pragma unroll
    for (int i = 0; i < 2; i++)
      #pragma unroll
      for (int j = 0; j < 4; j++)
        acc[i][j] = __builtin_amdgcn_wmma_f32_16x16x32_bf16(
            false, afr[i].v, false, bfr[j].v, (short)0, acc[i][j], false, false);
  }

  // ---------------------------- epilogue ------------------------------------
  const float scale = 0.088388347648318447f;   // 1/sqrt(128)
  #pragma unroll
  for (int j = 0; j < 4; j++) {
    int ncol = bn0 + wn * 64 + j * 16 + (lane & 15);
    float bval = (EPI == 0 || EPI == 1 || EPI == 3) ? bias[ncol] : 0.0f;
    #pragma unroll
    for (int i = 0; i < 2; i++) {
      int mbase = bm0 + wm * 32 + i * 16 + ((lane >> 4) * 8);
      #pragma unroll
      for (int r = 0; r < 8; r++) {
        int m = mbase + r;
        float v = acc[i][j][r] + bval;
        if (EPI == 1) {
          float partner = __shfl_xor(v, 1, 32);   // lane^1 holds column ncol^1
          if (ncol < 128) {
            int t = m & (Tdim - 1);
            // inv_freq = 10000^(-(n&~1)/128)
            float inv = __expf((float)(ncol & ~1) * -0.0719557841560639f);
            float th = (float)t * inv;
            float sn, cs; __sincosf(th, &sn, &cs);
            v = (ncol & 1) ? (v * cs + partner * sn) : (v * cs - partner * sn);
          }
        }
        if (EPI == 2) {
          v *= scale;
          if (ncol > m) v = -1e30f;               // causal mask
        }
        if (EPI == 0 || EPI == 2)
          ((float*)Cb)[(size_t)m * ldc + ncol] = v;
        else
          ((unsigned short*)Cb)[(size_t)m * ldc + ncol] = f32_to_bf16(v);
      }
    }
  }
}

// row softmax over T=2048, masked entries are -1e30; writes P = 2*softmax (bf16)
__global__ __launch_bounds__(256) void softmax_kernel(
    const float* __restrict__ S, unsigned short* __restrict__ P)
{
  const int tid = threadIdx.x;
  const size_t row = blockIdx.x;
  const float* s = S + row * (size_t)Tdim;
  unsigned short* p = P + row * (size_t)Tdim;

  float x[Tdim / 256];
  float mx = -3.0e38f;
  #pragma unroll
  for (int i = 0; i < Tdim / 256; i++) { x[i] = s[tid + i * 256]; mx = fmaxf(mx, x[i]); }
  #pragma unroll
  for (int o = 16; o > 0; o >>= 1) mx = fmaxf(mx, __shfl_xor(mx, o, 32));

  __shared__ float redm[8], reds[8];
  int wv = tid >> 5, ln = tid & 31;
  if (ln == 0) redm[wv] = mx;
  __syncthreads();
  mx = redm[0];
  #pragma unroll
  for (int w = 1; w < 8; w++) mx = fmaxf(mx, redm[w]);

  float sum = 0.f;
  #pragma unroll
  for (int i = 0; i < Tdim / 256; i++) { x[i] = __expf(x[i] - mx); sum += x[i]; }
  #pragma unroll
  for (int o = 16; o > 0; o >>= 1) sum += __shfl_xor(sum, o, 32);
  if (ln == 0) reds[wv] = sum;
  __syncthreads();
  sum = 0.f;
  #pragma unroll
  for (int w = 0; w < 8; w++) sum += reds[w];

  float k = 2.0f / sum;
  #pragma unroll
  for (int i = 0; i < Tdim / 256; i++) p[tid + i * 256] = f32_to_bf16(x[i] * k);
}

extern "C" void kernel_launch(void* const* d_in, const int* in_sizes, int n_in,
                              void* d_out, int out_size, void* d_ws, size_t ws_size,
                              hipStream_t stream) {
  (void)in_sizes; (void)n_in; (void)out_size; (void)ws_size;
  const float* X   = (const float*)d_in[0];
  const float* Wq  = (const float*)d_in[1];
  const float* bq  = (const float*)d_in[2];
  const float* Wk  = (const float*)d_in[3];
  const float* bk  = (const float*)d_in[4];
  const float* Wv  = (const float*)d_in[5];
  const float* bv  = (const float*)d_in[6];
  const float* Wfc = (const float*)d_in[7];
  const float* bfc = (const float*)d_in[8];

  const int Mrows = Bdim * Tdim;                       // 4096
  const size_t SZ_BTD_BF16 = (size_t)Bdim * Tdim * Ddim * 2;  // 16 MiB
  char* ws = (char*)d_ws;
  unsigned short* Qb = (unsigned short*)(ws);
  unsigned short* Kb = (unsigned short*)(ws + SZ_BTD_BF16);
  unsigned short* Vb = (unsigned short*)(ws + 2 * SZ_BTD_BF16);
  float*          S  = (float*)        (ws + 3 * SZ_BTD_BF16);
  unsigned short* P  = Qb;   // overlay: Q dead after scores
  unsigned short* O  = Kb;   // overlay: K dead after scores

  const long sBTD = (long)Tdim * Ddim;
  const long sBTT = (long)Tdim * Tdim;

  dim3 blk(256);
  dim3 gProj(Ddim / BN, Mrows / BM, 1);                // 16 x 32
  dim3 gAttn(Tdim / BN, Tdim / BM, Bdim);              // 16 x 16 x 2

  // 1) projections (A=f32 X, B=f32 W row-major [N][K] since Q = X @ W^T)
  gemm_kernel<false, false, false, 1><<<gProj, blk, 0, stream>>>(
      X, Wq, bq, Qb, Ddim, Ddim, Ddim, Ddim, 0, 0, 0);
  gemm_kernel<false, false, false, 1><<<gProj, blk, 0, stream>>>(
      X, Wk, bk, Kb, Ddim, Ddim, Ddim, Ddim, 0, 0, 0);
  gemm_kernel<false, false, false, 3><<<gProj, blk, 0, stream>>>(
      X, Wv, bv, Vb, Ddim, Ddim, Ddim, Ddim, 0, 0, 0);
  // 2) scores S = QK^T * 1/sqrt(128), causal masked (per batch)
  gemm_kernel<true, true, false, 2><<<gAttn, blk, 0, stream>>>(
      Qb, Kb, nullptr, S, Ddim, Ddim, Ddim, Tdim, sBTD, sBTD, sBTT);
  // 3) P = 2 * softmax(S) (bf16)
  softmax_kernel<<<dim3(Mrows), blk, 0, stream>>>(S, P);
  // 4) O = P @ V (B is [K][N] = V[s][d] -> NN path)
  gemm_kernel<true, true, true, 4><<<gAttn, blk, 0, stream>>>(
      P, Vb, nullptr, O, Tdim, Tdim, Ddim, Ddim, sBTT, sBTD, sBTD);
  // 5) out = O @ Wfc^T + bfc (f32 out)
  gemm_kernel<true, false, false, 0><<<gProj, blk, 0, stream>>>(
      O, Wfc, bfc, (float*)d_out, Ddim, Ddim, Ddim, Ddim, 0, 0, 0);
}
